// DenseSSMLayer_62302795596596
// MI455X (gfx1250) — matrix operand order repaired and checked by reading, not compiled, not run on hardware
//
#include <hip/hip_runtime.h>
#include <hip/hip_bf16.h>

// ---------------- types ----------------
typedef __attribute__((ext_vector_type(16))) _Float16 v16h;
typedef __attribute__((ext_vector_type(8)))  float    v8f;

union FragU {
    v16h v;
    unsigned int u[8];
};

#define BK 32
#define TB 256   // threads per block (8 wave32)

// GEMM epilogue modes
#define MODE_SIG 0   // sigmoid(acc + bias)            -> f32 out
#define MODE_LIN 1   // acc + bias                     -> f32 out
#define MODE_A   2   // tanh(acc+bias)*0.125, diag<-d  -> f16 out
#define MODE_OUT 3   // acc + D[col]*u[row,col]        -> f32 out

// ---------------- CDNA5 async global->LDS (ASYNCcnt) ----------------
// GLOBAL_LOAD_ASYNC_TO_LDS_B128: per-lane 16B, LDS address from VGPR (low 32
// bits of the flat shared pointer are the LDS byte offset per the aperture map).
__device__ __forceinline__ void async_copy16(const void* gptr, unsigned lds_off) {
    unsigned long long g = (unsigned long long)gptr;
    asm volatile("global_load_async_to_lds_b128 %0, %1, off"
                 :: "v"(lds_off), "v"(g) : "memory");
}
__device__ __forceinline__ unsigned lds_addr(const void* lptr) {
    return (unsigned)(unsigned long long)lptr;
}
__device__ __forceinline__ void wait_async0() {
    asm volatile("s_wait_asynccnt 0x0" ::: "memory");
}

// ---------------- f32 -> f16 convert ----------------
__global__ __launch_bounds__(256) void cvt_f32_f16(const float* __restrict__ in,
                                                   _Float16* __restrict__ out,
                                                   size_t n) {
    size_t i = (size_t)blockIdx.x * blockDim.x + threadIdx.x;
    if (i < n) out[i] = (_Float16)in[i];
}

// ---------------- unified WMMA GEMM: O = act( X[M,K] @ W[N,K]^T ) ----------------
// Block tile: (WM*MT*16) x (WN*NT*16), BK = 32, 8 wave32, K compile-time.
// Wave (wm,wn) owns MT x NT accumulators of 16x16; LDS double-buffered and
// filled by fully-unrolled async-to-LDS loads for tile k+1 while tile k is
// consumed by WMMAs.
template <int WM, int WN, int MT, int NT, int K, int MODE>
__global__ __launch_bounds__(TB) void gemm_f16(
    const _Float16* __restrict__ X,     // [M,K]
    const _Float16* __restrict__ W,     // [Ntot,K] (torch [out,in] layout)
    const float*    __restrict__ bias,  // [Ntot] or null
    int Ntot,
    float*    __restrict__ outF,        // MODE_SIG / MODE_LIN / MODE_OUT
    _Float16* __restrict__ outH,        // MODE_A
    const float* __restrict__ dbuf,     // MODE_A: diag replacement, [M,64]
    const float* __restrict__ Dvec,     // MODE_OUT: [Ntot]
    const float* __restrict__ uF)       // MODE_OUT: [M,Ntot] f32
{
    constexpr int BMt = WM * MT * 16;
    constexpr int BNt = WN * NT * 16;
    constexpr int CHX = (BMt * 4) / TB;   // 16B chunks per thread for X tile
    constexpr int CHW = (BNt * 4) / TB;   // 16B chunks per thread for W tile
    constexpr int NK  = K / BK;

    __shared__ _Float16 Xs[2][BMt * BK];
    __shared__ _Float16 Ws[2][BNt * BK];

    const int tid     = threadIdx.x;
    const int wave    = tid >> 5;
    const int lane    = tid & 31;
    const int wm      = wave / WN;
    const int wn      = wave % WN;
    const int rowBase = blockIdx.x * BMt;
    const int colBase = blockIdx.y * BNt;
    const int n       = lane & 15;
    const int hi      = lane >> 4;

    // hoist per-thread staging addresses (global base + both LDS buffers)
    const _Float16* xg[CHX];
    unsigned        xl[CHX][2];
#pragma unroll
    for (int i = 0; i < CHX; ++i) {
        const int idx = tid + i * TB, r = idx >> 2, c = idx & 3;
        xg[i]    = X + (size_t)(rowBase + r) * K + c * 8;
        xl[i][0] = lds_addr(&Xs[0][r * BK + c * 8]);
        xl[i][1] = lds_addr(&Xs[1][r * BK + c * 8]);
    }
    const _Float16* wg[CHW];
    unsigned        wl[CHW][2];
#pragma unroll
    for (int i = 0; i < CHW; ++i) {
        const int idx = tid + i * TB, r = idx >> 2, c = idx & 3;
        wg[i]    = W + (size_t)(colBase + r) * K + c * 8;
        wl[i][0] = lds_addr(&Ws[0][r * BK + c * 8]);
        wl[i][1] = lds_addr(&Ws[1][r * BK + c * 8]);
    }

    v8f acc[MT][NT] = {};

    // issue one K-tile of async copies into LDS buffer `buf` (straight-line)
    auto load_tile = [&](int k0, int buf) {
#pragma unroll
        for (int i = 0; i < CHX; ++i) async_copy16(xg[i] + k0, xl[i][buf]);
#pragma unroll
        for (int i = 0; i < CHW; ++i) async_copy16(wg[i] + k0, wl[i][buf]);
    };

    load_tile(0, 0);
    wait_async0();
    __syncthreads();

    for (int kk = 0; kk < NK; ++kk) {
        const int cur = kk & 1;
        if (kk + 1 < NK) load_tile((kk + 1) * BK, cur ^ 1);

        const unsigned int* Xu = (const unsigned int*)Xs[cur];
        const unsigned int* Wu = (const unsigned int*)Ws[cur];

        // A fragments: 16x32 (lane m = lane&15; K = hi*16 + 2v,2v+1, matching B)
        FragU af[MT];
#pragma unroll
        for (int mt = 0; mt < MT; ++mt) {
            const int mrow = (wm * MT + mt) * 16 + n;
#pragma unroll
            for (int v = 0; v < 8; ++v) {
                const int kb = 2 * v + hi * 16;  // even
                af[mt].u[v] = Xu[mrow * (BK / 2) + (kb >> 1)];
            }
        }
#pragma unroll
        for (int nt = 0; nt < NT; ++nt) {
            FragU bf;                      // column of W-tile; K = hi*16 + 2v,2v+1
            const int ncol = (wn * NT + nt) * 16 + n;
#pragma unroll
            for (int v = 0; v < 8; ++v)
                bf.u[v] = Wu[ncol * (BK / 2) + hi * 8 + v];
#pragma unroll
            for (int mt = 0; mt < MT; ++mt)
                acc[mt][nt] = __builtin_amdgcn_wmma_f32_16x16x32_f16(
                    false, af[mt].v, false, bf.v, (short)0, acc[mt][nt], false, false);
        }

        wait_async0();       // next tile fully landed in LDS
        __syncthreads();
    }

    // epilogue: C/D layout -> element (r, lane): m = r + 8*hi, n = lane&15
#pragma unroll
    for (int mt = 0; mt < MT; ++mt) {
#pragma unroll
        for (int nt = 0; nt < NT; ++nt) {
            const int gcol = colBase + (wn * NT + nt) * 16 + n;
#pragma unroll
            for (int r = 0; r < 8; ++r) {
                const int grow = rowBase + (wm * MT + mt) * 16 + r + 8 * hi;
                const float a = acc[mt][nt][r];
                const size_t oidx = (size_t)grow * Ntot + gcol;
                if constexpr (MODE == MODE_SIG) {
                    const float x = a + bias[gcol];
                    outF[oidx] = 1.0f / (1.0f + expf(-x));
                } else if constexpr (MODE == MODE_LIN) {
                    outF[oidx] = a + bias[gcol];
                } else if constexpr (MODE == MODE_A) {
                    float x = tanhf(a + bias[gcol]) * 0.125f;   // 1/sqrt(64)
                    const int ii = gcol >> 6;
                    if (ii == (gcol & 63)) x = dbuf[(size_t)grow * 64 + ii];
                    outH[oidx] = (_Float16)x;
                } else {  // MODE_OUT
                    outF[oidx] = a + Dvec[gcol] * uF[oidx];
                }
            }
        }
    }
}

// ---------------- sequential scan: h_t = Ad_t h_{t-1} + Bu_t ----------------
// One block per batch; 64x64 f16 tile double-buffered in LDS; 4 lanes per row,
// intra-wave __shfl_xor reduction; next tile prefetched into registers.
__global__ __launch_bounds__(TB) void scan_kernel(
    const _Float16* __restrict__ Ad,   // [B*T, 4096]
    const float*    __restrict__ bu,   // [B*T, 64]
    _Float16*       __restrict__ hs,   // [B*T, 64]  (f16 for the output GEMM)
    int T)
{
    __shared__ _Float16 Atile[2][64 * 64];   // 2 x 8KB
    __shared__ float h[64];

    const int tid = threadIdx.x;
    const size_t base = (size_t)blockIdx.x * T;

    if (tid < 64) h[tid] = 0.0f;

    // preload tile 0: 8KB = 256 threads x 32B
    {
        const uint4* src = (const uint4*)(Ad + base * 4096) + tid * 2;
        uint4* dst = (uint4*)(Atile[0]) + tid * 2;
        dst[0] = src[0];
        dst[1] = src[1];
    }
    __syncthreads();

    const int i = tid >> 2;        // output row
    const int q = tid & 3;         // 16-wide column segment

    for (int t = 0; t < T; ++t) {
        const int cur = t & 1;
        uint4 r0, r1;
        if (t + 1 < T) {           // issue next-tile loads early (latency overlap)
            const uint4* src = (const uint4*)(Ad + (base + t + 1) * 4096) + tid * 2;
            r0 = src[0];
            r1 = src[1];
        }
        if (t + 2 < T)             // keep the t+2 line inbound (global_prefetch_b8)
            __builtin_prefetch(Ad + (base + t + 2) * 4096 + tid * 16, 0, 2);

        const _Float16* Arow = Atile[cur] + i * 64 + q * 16;
        float s = 0.0f;
#pragma unroll
        for (int j = 0; j < 16; ++j)
            s += (float)Arow[j] * h[q * 16 + j];
        s += __shfl_xor(s, 1);
        s += __shfl_xor(s, 2);

        float hn = 0.0f;
        if (q == 0) hn = s + bu[(base + t) * 64 + i];

        __syncthreads();           // all reads of h / Atile[cur] done

        if (q == 0) {
            h[i] = hn;
            hs[(base + t) * 64 + i] = (_Float16)hn;
        }
        if (t + 1 < T) {
            uint4* dst = (uint4*)(Atile[cur ^ 1]) + tid * 2;
            dst[0] = r0;
            dst[1] = r1;
        }
        __syncthreads();           // writes visible for next iteration
    }
}

// ---------------- host-side orchestration ----------------
static inline size_t align256(size_t x) { return (x + 255) & ~(size_t)255; }

extern "C" void kernel_launch(void* const* d_in, const int* in_sizes, int n_in,
                              void* d_out, int out_size, void* d_ws, size_t ws_size,
                              hipStream_t stream) {
    (void)in_sizes; (void)n_in; (void)out_size; (void)ws_size;

    const int Bn = 8, T = 2048, DM = 512, N = 64;
    const int M = Bn * T;                       // 16384

    const float* u    = (const float*)d_in[0];  // [B,T,512]
    const float* Wd_w = (const float*)d_in[1];  // [64,512]
    const float* Wd_b = (const float*)d_in[2];  // [64]
    const float* WA_w = (const float*)d_in[3];  // [4096,512]
    const float* WA_b = (const float*)d_in[4];  // [4096]
    const float* WB_w = (const float*)d_in[5];  // [64,512]
    const float* WB_b = (const float*)d_in[6];  // [64]
    const float* Cw   = (const float*)d_in[7];  // [512,64]
    const float* Dv   = (const float*)d_in[8];  // [512]
    float* out = (float*)d_out;                 // [B,T,512]

    // workspace carve-up
    char* w = (char*)d_ws;
    size_t off = 0;
    _Float16* u_h  = (_Float16*)(w + off); off = align256(off + (size_t)M * DM * 2);
    _Float16* WA_h = (_Float16*)(w + off); off = align256(off + (size_t)N * N * DM * 2);
    _Float16* Wd_h = (_Float16*)(w + off); off = align256(off + (size_t)N * DM * 2);
    _Float16* WB_h = (_Float16*)(w + off); off = align256(off + (size_t)N * DM * 2);
    _Float16* Cw_h = (_Float16*)(w + off); off = align256(off + (size_t)DM * N * 2);
    float*    dbuf = (float*)(w + off);    off = align256(off + (size_t)M * N * 4);
    float*    bubuf= (float*)(w + off);    off = align256(off + (size_t)M * N * 4);
    _Float16* Ad_h = (_Float16*)(w + off); off = align256(off + (size_t)M * N * N * 2);
    _Float16* hs_h = (_Float16*)(w + off); off = align256(off + (size_t)M * N * 2);

    // 1) f32 -> f16 converts
    {
        size_t n;
        n = (size_t)M * DM;
        cvt_f32_f16<<<(unsigned)((n + 255) / 256), 256, 0, stream>>>(u, u_h, n);
        n = (size_t)N * N * DM;
        cvt_f32_f16<<<(unsigned)((n + 255) / 256), 256, 0, stream>>>(WA_w, WA_h, n);
        n = (size_t)N * DM;
        cvt_f32_f16<<<(unsigned)((n + 255) / 256), 256, 0, stream>>>(Wd_w, Wd_h, n);
        cvt_f32_f16<<<(unsigned)((n + 255) / 256), 256, 0, stream>>>(WB_w, WB_h, n);
        n = (size_t)DM * N;
        cvt_f32_f16<<<(unsigned)((n + 255) / 256), 256, 0, stream>>>(Cw, Cw_h, n);
    }

    // 2) d = sigmoid(u Wd^T + b)   [16384 x 64], K=512 (128x64 tile, 8x1 waves)
    gemm_f16<8, 1, 1, 4, 512, MODE_SIG><<<dim3(M / 128, 1), TB, 0, stream>>>(
        u_h, Wd_h, Wd_b, N, dbuf, nullptr, nullptr, nullptr, nullptr);

    // 3) Bu = u WB^T + b           [16384 x 64], K=512
    gemm_f16<8, 1, 1, 4, 512, MODE_LIN><<<dim3(M / 128, 1), TB, 0, stream>>>(
        u_h, WB_h, WB_b, N, bubuf, nullptr, nullptr, nullptr, nullptr);

    // 4) Ad = tanh(u WA^T + b)/8, diag <- d   [16384 x 4096], K=512
    //    128x128 tile, 4x2 waves, 8 WMMA per wave per K-step
    gemm_f16<4, 2, 2, 4, 512, MODE_A><<<dim3(M / 128, (N * N) / 128), TB, 0, stream>>>(
        u_h, WA_h, WA_b, N * N, nullptr, Ad_h, dbuf, nullptr, nullptr);

    // 5) recurrence, one block per batch
    scan_kernel<<<Bn, TB, 0, stream>>>(Ad_h, bubuf, hs_h, T);

    // 6) out = hs C^T + D*u        [16384 x 512], K=64
    gemm_f16<4, 2, 2, 4, 64, MODE_OUT><<<dim3(M / 128, DM / 128), TB, 0, stream>>>(
        hs_h, Cw_h, nullptr, DM, out, nullptr, nullptr, Dv, u);
}
